// _EventReasoningModule_69166153335009
// MI455X (gfx1250) — matrix operand order repaired and checked by reading, not compiled, not run on hardware
//
#include <hip/hip_runtime.h>

typedef __attribute__((ext_vector_type(16))) _Float16 v16h;
typedef __attribute__((ext_vector_type(8)))  float    v8f;

#define Gc    256
#define PLMc  768
#define BSc   16384     // B*S tokens
#define NWMc  50000
#define EWMc  800000
#define EFc   131072

// ---------------- graph / elementwise kernels ----------------

__global__ void k_scatter_add(const float* __restrict__ h, const int* __restrict__ src,
                              const int* __restrict__ dst, float* __restrict__ m, int E) {
  int e = blockIdx.x;
  if (e >= E) return;
  int s = src[e], d = dst[e];
  int g = threadIdx.x;
  atomicAdd(&m[(size_t)d * Gc + g], h[(size_t)s * Gc + g]);
}

__global__ void k_add(const float* __restrict__ a, const float* __restrict__ b,
                      float* __restrict__ c, size_t n) {
  size_t i = (size_t)blockIdx.x * blockDim.x + threadIdx.x;
  if (i < n) c[i] = a[i] + b[i];
}

__global__ void k_tmp(const float* __restrict__ hwm, const float* __restrict__ extra,
                      const int* __restrict__ t2n, float* __restrict__ tmp) {
  int i = blockIdx.x;
  int g = threadIdx.x;
  float acc = 0.f;
#pragma unroll
  for (int k = 0; k < 4; ++k) {
    int node = t2n[i * 4 + k] + 2;
    acc += (node < 2) ? extra[(size_t)node * Gc + g]
                      : hwm[(size_t)(node - 2) * Gc + g];
  }
  tmp[(size_t)i * Gc + g] = acc;
}

__global__ void k_gather_rows(const float* __restrict__ srcM, const int* __restrict__ ids,
                              float* __restrict__ dstM) {
  int i = blockIdx.x;
  int g = threadIdx.x;
  dstM[(size_t)i * Gc + g] = srcM[(size_t)ids[i] * Gc + g];
}

// ---------------- weight pre-pack: f32 [Ktot x N] -> f16 in wave32 B-fragment order ----
// Wp[(((k/32) * (N/16) + n/16) * 32 + lane) * 16 + j],
//   lane = (n%16) + 16*(((k%32)/8)%2), j = 8*((k%32)/16) + k%8   (ISA 7.12.2 B 32x16 16-bit)
__global__ void k_packW(const float* __restrict__ W, _Float16* __restrict__ Wp,
                        int Ktot, int N) {
  int idx = blockIdx.x * 256 + threadIdx.x;
  if (idx >= Ktot * N) return;
  int j    = idx & 15;
  int lane = (idx >> 4) & 31;
  int rest = idx >> 9;
  int nt16 = N >> 4;
  int ct   = rest % nt16;
  int kb   = rest / nt16;
  int k = kb * 32 + (j >> 3) * 16 + (lane >> 4) * 8 + (j & 7);
  int n = ct * 16 + (lane & 15);
  Wp[idx] = (_Float16)W[(size_t)k * N + n];
}

// ---------------- WMMA GEMM helpers ----------------

__device__ __forceinline__ v16h loadA16(const float* ap) {
  // ISA A 16x32 16-bit layout: per lane two runs of 8 consecutive K values.
  float4 q0 = *(const float4*)(ap);
  float4 q1 = *(const float4*)(ap + 4);
  float4 q2 = *(const float4*)(ap + 16);
  float4 q3 = *(const float4*)(ap + 20);
  v16h a;
  a[0]  = (_Float16)q0.x; a[1]  = (_Float16)q0.y;
  a[2]  = (_Float16)q0.z; a[3]  = (_Float16)q0.w;
  a[4]  = (_Float16)q1.x; a[5]  = (_Float16)q1.y;
  a[6]  = (_Float16)q1.z; a[7]  = (_Float16)q1.w;
  a[8]  = (_Float16)q2.x; a[9]  = (_Float16)q2.y;
  a[10] = (_Float16)q2.z; a[11] = (_Float16)q2.w;
  a[12] = (_Float16)q3.x; a[13] = (_Float16)q3.y;
  a[14] = (_Float16)q3.z; a[15] = (_Float16)q3.w;
  return a;
}

// Software-pipelined per-segment K loop: fragments for block i+1 are loaded
// before the 4 WMMAs of block i are issued, hiding load latency.
__device__ __forceinline__ void gemm_seg(const float* arow, const _Float16* bp,
                                         size_t bstride, int nb, v8f* c) {
  v16h a  = loadA16(arow);
  v16h b0 = *(const v16h*)(bp);
  v16h b1 = *(const v16h*)(bp + 512);
  v16h b2 = *(const v16h*)(bp + 1024);
  v16h b3 = *(const v16h*)(bp + 1536);
  for (int i = 0; i < nb - 1; ++i) {
    arow += 32;
    bp   += bstride;
    v16h an = loadA16(arow);
    v16h n0 = *(const v16h*)(bp);
    v16h n1 = *(const v16h*)(bp + 512);
    v16h n2 = *(const v16h*)(bp + 1024);
    v16h n3 = *(const v16h*)(bp + 1536);
    c[0] = __builtin_amdgcn_wmma_f32_16x16x32_f16(false, a, false, b0, (short)0, c[0], false, false);
    c[1] = __builtin_amdgcn_wmma_f32_16x16x32_f16(false, a, false, b1, (short)0, c[1], false, false);
    c[2] = __builtin_amdgcn_wmma_f32_16x16x32_f16(false, a, false, b2, (short)0, c[2], false, false);
    c[3] = __builtin_amdgcn_wmma_f32_16x16x32_f16(false, a, false, b3, (short)0, c[3], false, false);
    a = an; b0 = n0; b1 = n1; b2 = n2; b3 = n3;
  }
  c[0] = __builtin_amdgcn_wmma_f32_16x16x32_f16(false, a, false, b0, (short)0, c[0], false, false);
  c[1] = __builtin_amdgcn_wmma_f32_16x16x32_f16(false, a, false, b1, (short)0, c[1], false, false);
  c[2] = __builtin_amdgcn_wmma_f32_16x16x32_f16(false, a, false, b2, (short)0, c[2], false, false);
  c[3] = __builtin_amdgcn_wmma_f32_16x16x32_f16(false, a, false, b3, (short)0, c[3], false, false);
}

// C[M x N] = act( concat_K(A0|A1|A2)[M x Ktot] @ W + bias )
// Block = 256 thr = 8 waves; wave tile 16(M) x 64(N); block tile 128 x 64.
// Out-of-range rows are clamped (their D rows are garbage but never stored).
__global__ __launch_bounds__(256) void k_gemm_wmma(
    const float* A0, int K0, const float* A1, int K1, const float* A2, int K2,
    const _Float16* __restrict__ Wp, const float* __restrict__ bias,
    float* __restrict__ C, int M, int N, int relu) {
  const int lane = threadIdx.x & 31;
  const int wave = threadIdx.x >> 5;
  const int hr   = lane & 15;
  const int hi   = lane >> 4;
  const int row  = blockIdx.y * 128 + wave * 16 + hr;
  const int arow = (row < M) ? row : (M - 1);          // clamp, no in-loop guard
  const int ct0  = (blockIdx.x * 64) >> 4;
  const int nt16 = N >> 4;
  const size_t bstride = (size_t)nt16 * 512;           // halfs per 32-K block
  const _Float16* bp0 = Wp + ((size_t)ct0 * 32 + lane) * 16;

  v8f c[4] = {};

  gemm_seg(A0 + (size_t)arow * K0 + hi * 8, bp0, bstride, K0 >> 5, c);
  if (K1)
    gemm_seg(A1 + (size_t)arow * K1 + hi * 8,
             bp0 + (size_t)(K0 >> 5) * bstride, bstride, K1 >> 5, c);
  if (K2)
    gemm_seg(A2 + (size_t)arow * K2 + hi * 8,
             bp0 + (size_t)((K0 + K1) >> 5) * bstride, bstride, K2 >> 5, c);

  if (row >= M) return;
  const int rbase = blockIdx.y * 128 + wave * 16;
#pragma unroll
  for (int u = 0; u < 4; ++u) {
    int col = blockIdx.x * 64 + u * 16 + hr;
    float bv = bias[col];
#pragma unroll
    for (int i = 0; i < 8; ++i) {
      int r = rbase + i + hi * 8;
      if (r < M) {
        float v = c[u][i] + bv;
        if (relu) v = fmaxf(v, 0.f);
        C[(size_t)r * N + col] = v;
      }
    }
  }
}

// ---------------- driver ----------------
extern "C" void kernel_launch(void* const* d_in, const int* in_sizes, int n_in,
                              void* d_out, int out_size, void* d_ws, size_t ws_size,
                              hipStream_t stream) {
  const float* text   = (const float*)d_in[0];
  const float* wm_x   = (const float*)d_in[1];
  const int*   wm_e   = (const int*)d_in[2];
  const int*   t2n    = (const int*)d_in[3];
  const int*   f_ids  = (const int*)d_in[4];
  const int*   f_e    = (const int*)d_in[5];
  const float* extra  = (const float*)d_in[6];
  const float* wm_W1  = (const float*)d_in[7];
  const float* wm_b1  = (const float*)d_in[8];
  const float* wm_W2  = (const float*)d_in[9];
  const float* wm_b2  = (const float*)d_in[10];
  const float* f_W1   = (const float*)d_in[11];
  const float* f_b1   = (const float*)d_in[12];
  const float* f_W2   = (const float*)d_in[13];
  const float* f_b2   = (const float*)d_in[14];
  const float* fc1_W  = (const float*)d_in[15];
  const float* fc1_b  = (const float*)d_in[16];
  const float* fc3_W  = (const float*)d_in[17];
  const float* fc3_b  = (const float*)d_in[18];
  float* out = (float*)d_out;

  float* ws = (float*)d_ws;
  const size_t NG = (size_t)NWMc * Gc;
  const size_t TG = (size_t)BSc * Gc;
  float* h_wm = ws;
  float* mbuf = h_wm + NG;
  float* abuf = mbuf + NG;
  float* tmp  = abuf + NG;
  float* gte  = tmp + TG;
  float* xg   = gte + TG;
  _Float16* wpWM1 = (_Float16*)(xg + TG);
  _Float16* wpWM2 = wpWM1 + Gc * Gc;
  _Float16* wpFC1 = wpWM2 + Gc * Gc;
  _Float16* wpF1  = wpFC1 + (Gc + PLMc) * Gc;
  _Float16* wpF2  = wpF1 + Gc * Gc;
  _Float16* wpFC3 = wpF2 + Gc * Gc;

  dim3 b256(256);

  // ===== pre-pack all weights to f16 B-fragment layout =====
  k_packW<<<(Gc * Gc + 255) / 256, b256, 0, stream>>>(wm_W1, wpWM1, Gc, Gc);
  k_packW<<<(Gc * Gc + 255) / 256, b256, 0, stream>>>(wm_W2, wpWM2, Gc, Gc);
  k_packW<<<((Gc + PLMc) * Gc + 255) / 256, b256, 0, stream>>>(fc1_W, wpFC1, Gc + PLMc, Gc);
  k_packW<<<(Gc * Gc + 255) / 256, b256, 0, stream>>>(f_W1, wpF1, Gc, Gc);
  k_packW<<<(Gc * Gc + 255) / 256, b256, 0, stream>>>(f_W2, wpF2, Gc, Gc);
  k_packW<<<((PLMc + 2 * Gc) * PLMc + 255) / 256, b256, 0, stream>>>(fc3_W, wpFC3,
                                                                     PLMc + 2 * Gc, PLMc);

  // ===== WM GNN layer 1 =====
  hipMemsetAsync(mbuf, 0, NG * sizeof(float), stream);
  k_scatter_add<<<EWMc, b256, 0, stream>>>(wm_x, wm_e, wm_e + EWMc, mbuf, EWMc);
  k_add<<<(unsigned)((NG + 255) / 256), b256, 0, stream>>>(wm_x, mbuf, abuf, NG);
  {
    dim3 grid(Gc / 64, (NWMc + 127) / 128);
    k_gemm_wmma<<<grid, b256, 0, stream>>>(abuf, Gc, nullptr, 0, nullptr, 0,
                                           wpWM1, wm_b1, h_wm, NWMc, Gc, 1);
  }
  // ===== WM GNN layer 2 =====
  hipMemsetAsync(mbuf, 0, NG * sizeof(float), stream);
  k_scatter_add<<<EWMc, b256, 0, stream>>>(h_wm, wm_e, wm_e + EWMc, mbuf, EWMc);
  k_add<<<(unsigned)((NG + 255) / 256), b256, 0, stream>>>(h_wm, mbuf, abuf, NG);
  {
    dim3 grid(Gc / 64, (NWMc + 127) / 128);
    k_gemm_wmma<<<grid, b256, 0, stream>>>(abuf, Gc, nullptr, 0, nullptr, 0,
                                           wpWM2, wm_b2, h_wm, NWMc, Gc, 1);
  }
  // ===== token gather-sum =====
  k_tmp<<<BSc, b256, 0, stream>>>(h_wm, extra, t2n, tmp);
  // ===== fc1: gte = [tmp | text] @ fc1_W + fc1_b (no relu) =====
  {
    dim3 grid(Gc / 64, BSc / 128);
    k_gemm_wmma<<<grid, b256, 0, stream>>>(tmp, Gc, text, PLMc, nullptr, 0,
                                           wpFC1, fc1_b, gte, BSc, Gc, 0);
  }
  // ===== gather rows for FSTM =====
  k_gather_rows<<<BSc, b256, 0, stream>>>(gte, f_ids, xg);
  // ===== FSTM layer 1 =====
  hipMemsetAsync(mbuf, 0, TG * sizeof(float), stream);
  k_scatter_add<<<EFc, b256, 0, stream>>>(xg, f_e, f_e + EFc, mbuf, EFc);
  k_add<<<(unsigned)((TG + 255) / 256), b256, 0, stream>>>(xg, mbuf, abuf, TG);
  {
    dim3 grid(Gc / 64, BSc / 128);
    k_gemm_wmma<<<grid, b256, 0, stream>>>(abuf, Gc, nullptr, 0, nullptr, 0,
                                           wpF1, f_b1, xg, BSc, Gc, 1);
  }
  // ===== FSTM layer 2 =====
  hipMemsetAsync(mbuf, 0, TG * sizeof(float), stream);
  k_scatter_add<<<EFc, b256, 0, stream>>>(xg, f_e, f_e + EFc, mbuf, EFc);
  k_add<<<(unsigned)((TG + 255) / 256), b256, 0, stream>>>(xg, mbuf, abuf, TG);
  {
    dim3 grid(Gc / 64, BSc / 128);
    k_gemm_wmma<<<grid, b256, 0, stream>>>(abuf, Gc, nullptr, 0, nullptr, 0,
                                           wpF2, f_b2, xg, BSc, Gc, 1);
  }
  // ===== fc3: out = [text | tmp | temporal] @ fc3_W + fc3_b =====
  {
    dim3 grid(PLMc / 64, BSc / 128);
    k_gemm_wmma<<<grid, b256, 0, stream>>>(text, PLMc, tmp, Gc, xg, Gc,
                                           wpFC3, fc3_b, out, BSc, PLMc, 0);
  }
}
